// Route_GNN_76149770158376
// MI455X (gfx1250) — matrix operand order repaired and checked by reading, not compiled, not run on hardware
//
#include <hip/hip_runtime.h>
#include <hip/hip_bf16.h>

typedef __attribute__((ext_vector_type(16))) __bf16 v16bf;
typedef __attribute__((ext_vector_type(8)))  float  v8f;

// ---------------- degree / normalization ----------------
__global__ void k_zero_deg(int* __restrict__ deg, int n) {
  int i = blockIdx.x * blockDim.x + threadIdx.x;
  if (i < n) deg[i] = 0;
}

__global__ void k_count_deg(const int* __restrict__ dst, int E, int* __restrict__ deg) {
  int i = blockIdx.x * blockDim.x + threadIdx.x;
  if (i < E) atomicAdd(&deg[dst[i]], 1);
}

__global__ void k_dinv(const int* __restrict__ deg, float* __restrict__ dinv, int n) {
  int i = blockIdx.x * blockDim.x + threadIdx.x;
  if (i < n) dinv[i] = rsqrtf((float)(deg[i] + 1));  // +1 for self loop
}

// ---------------- weight pre-swizzle into per-lane WMMA B order ----------------
// Bsw layout: [nt][kb][lane][16 bf16], lane's 16 elements contiguous (32B).
// Element i of lane: K = kb*32 + (lane>=16 ? 16 : 0) + i, N = nt*16 + (lane&15).
// Out-of-range columns (Nw=8 layer) stored as zero.
__global__ void k_swizzleW(const float* __restrict__ W, __bf16* __restrict__ Bsw,
                           int K, int Nw, int NTN) {
  int idx = blockIdx.x * blockDim.x + threadIdx.x;  // (nt*nKB + kb)*32 + lane
  int nKB = K >> 5;
  if (idx >= NTN * nKB * 32) return;
  int lane = idx & 31;
  int kb = (idx >> 5) % nKB;
  int nt = (idx >> 5) / nKB;
  int half = lane >> 4, l16 = lane & 15;
  int n = nt * 16 + l16;
  int kbase = kb * 32 + half * 16;
  __bf16* o = Bsw + (size_t)idx * 16;
#pragma unroll
  for (int i = 0; i < 16; ++i) {
    float w = (n < Nw) ? W[(size_t)(kbase + i) * Nw + n] : 0.0f;
    o[i] = (__bf16)w;
  }
}

// ---------------- WMMA GEMM: T[M,Nw] = X[M,K] * W[K,Nw] ----------------
// One wave computes a 16 x (NTN*16) output block: A loaded once per k-step,
// NTN independent WMMAs per k-step. K multiple of 32.
template <int NTN>
__global__ void k_gemm_wmma(const float* __restrict__ X,
                            const __bf16* __restrict__ Bsw,
                            float* __restrict__ T,
                            int M, int K, int Nw, int mTiles) {
  int wave = threadIdx.x >> 5;
  int mt = blockIdx.x * (blockDim.x >> 5) + wave;
  if (mt >= mTiles) return;  // wave-uniform; EXEC all-ones for WMMA

  int lane = threadIdx.x & 31;
  int half = lane >> 4;
  int l16  = lane & 15;
  int m = mt * 16 + l16;
  if (m >= M) m = M - 1;  // clamped read (M is tile-aligned here anyway)
  const float* arow = X + (size_t)m * K;

  v8f zero = {};
  v8f acc[NTN];
#pragma unroll
  for (int t = 0; t < NTN; ++t) acc[t] = zero;

  const int nKB = K >> 5;
  for (int kb = 0; kb < nKB; ++kb) {
    // A 16x32 bf16: elems 0..15 -> K = kb*32 + half*16 + i
    int ka = kb * 32 + half * 16;
    v16bf a;
#pragma unroll
    for (int i = 0; i < 8; ++i) {
      a[i]     = (__bf16)arow[ka + i];
      a[i + 8] = (__bf16)arow[ka + 8 + i];
    }
#pragma unroll
    for (int t = 0; t < NTN; ++t) {
      const v16bf b = *(const v16bf*)(Bsw + (((size_t)t * nKB + kb) * 32 + lane) * 16);
      acc[t] = __builtin_amdgcn_wmma_f32_16x16x32_bf16(
          /*neg_a=*/false, a, /*neg_b=*/false, b,
          /*c_mod=*/(short)0, acc[t], /*reuse_a=*/false, /*reuse_b=*/false);
    }
  }

  // C/D layout: VGPR r -> M = mt*16 + half*8 + r, N = nt*16 + l16
  int mbase = mt * 16 + half * 8;
#pragma unroll
  for (int t = 0; t < NTN; ++t) {
    int n = t * 16 + l16;
    if (n < Nw) {
#pragma unroll
      for (int r = 0; r < 8; ++r) {
        int mm = mbase + r;
        if (mm < M) T[(size_t)mm * Nw + n] = acc[t][r];
      }
    }
  }
}

// ---------------- aggregation ----------------
// agg[v,f] = dinv[v]^2 * t[v,f]  (self-loop term; also zero-initializes agg)
__global__ void k_selfinit(const float* __restrict__ t, const float* __restrict__ dinv,
                           float* __restrict__ agg, int n, int F) {
  int chunks = F >> 2;
  int i = blockIdx.x * blockDim.x + threadIdx.x;
  if (i >= n * chunks) return;
  int v = i / chunks;
  int f = (i - v * chunks) << 2;
  float w = dinv[v];
  w *= w;
  const float4 x = *(const float4*)(t + (size_t)v * F + f);
  float4 y;
  y.x = w * x.x; y.y = w * x.y; y.z = w * x.z; y.w = w * x.w;
  *(float4*)(agg + (size_t)v * F + f) = y;
}

// agg[d] += dinv[s]*dinv[d] * t[s]  over all edges; float4 gather + 4 atomics
__global__ void k_scatter(const int* __restrict__ src, const int* __restrict__ dst,
                          const float* __restrict__ dinv, const float* __restrict__ t,
                          float* __restrict__ agg, int E, int F) {
  int chunks = F >> 2;
  long long i = (long long)blockIdx.x * blockDim.x + threadIdx.x;
  if (i >= (long long)E * chunks) return;
  int e = (int)(i / chunks);
  int f = (int)(i - (long long)e * chunks) << 2;
  int s = src[e], d = dst[e];
  float w = dinv[s] * dinv[d];
  const float4 x = *(const float4*)(t + (size_t)s * F + f);
  float* o = agg + (size_t)d * F + f;
  atomicAdd(o + 0, w * x.x);
  atomicAdd(o + 1, w * x.y);
  atomicAdd(o + 2, w * x.z);
  atomicAdd(o + 3, w * x.w);
}

__global__ void k_bias_relu(float* __restrict__ h, const float* __restrict__ b,
                            int n, int F) {
  int i = blockIdx.x * blockDim.x + threadIdx.x;
  if (i >= n * F) return;
  int f = i & (F - 1);  // F is a power of two (128 or 8)
  float v = h[i] + b[f];
  h[i] = v > 0.0f ? v : 0.0f;
}

// ---------------- dense head: out[v,c] = h[v,:8] . Wc[:,c] + bc[c] ----------------
__global__ void k_final(const float* __restrict__ h, const float* __restrict__ Wc,
                        const float* __restrict__ bc, float* __restrict__ out, int n) {
  int i = blockIdx.x * blockDim.x + threadIdx.x;
  if (i >= n * 16) return;
  int v = i >> 4, c = i & 15;
  const float* hr = h + (size_t)v * 8;
  float acc = bc[c];
#pragma unroll
  for (int j = 0; j < 8; ++j) acc += hr[j] * Wc[j * 16 + c];
  out[i] = acc;
}

// ---------------- driver ----------------
extern "C" void kernel_launch(void* const* d_in, const int* in_sizes, int n_in,
                              void* d_out, int out_size, void* d_ws, size_t ws_size,
                              hipStream_t stream) {
  const float* x  = (const float*)d_in[0];
  const int*   ei = (const int*)d_in[1];
  const float* W1 = (const float*)d_in[2];
  const float* b1 = (const float*)d_in[3];
  const float* W2 = (const float*)d_in[4];
  const float* b2 = (const float*)d_in[5];
  const float* W3 = (const float*)d_in[6];
  const float* b3 = (const float*)d_in[7];
  const float* Wc = (const float*)d_in[8];
  const float* bc = (const float*)d_in[9];

  const int F = 128, F3 = 8, C = 16;
  const int N = in_sizes[0] / F;
  const int E = in_sizes[1] / 2;
  const int* src = ei;
  const int* dst = ei + E;

  // workspace layout
  float*  t    = (float*)d_ws;                  // N*128 GEMM output
  float*  bufA = t    + (size_t)N * F;          // N*128
  float*  bufB = bufA + (size_t)N * F;          // N*128
  int*    deg  = (int*)(bufB + (size_t)N * F);  // N
  float*  dinv = (float*)(deg + N);             // N
  __bf16* Wsw  = (__bf16*)(dinv + N);           // up to 8*4*32*16 bf16 = 32KB

  auto cdiv = [](long long a, int b) { return (int)((a + b - 1) / b); };

  k_zero_deg<<<cdiv(N, 256), 256, 0, stream>>>(deg, N);
  k_count_deg<<<cdiv(E, 256), 256, 0, stream>>>(dst, E, deg);
  k_dinv<<<cdiv(N, 256), 256, 0, stream>>>(deg, dinv, N);

  const int mTiles = (N + 15) / 16;

  auto layer = [&](const float* in, const float* W, const float* b,
                   float* tb, float* ab, int Fin, int Fout, bool wide) {
    int ntN = (Fout + 15) / 16;
    int nKB = Fin >> 5;
    int swTotal = ntN * nKB * 32;
    k_swizzleW<<<cdiv(swTotal, 256), 256, 0, stream>>>(W, Wsw, Fin, Fout, ntN);
    if (wide) {
      k_gemm_wmma<8><<<cdiv(mTiles, 4), 128, 0, stream>>>(in, Wsw, tb, N, Fin, Fout, mTiles);
    } else {
      k_gemm_wmma<1><<<cdiv(mTiles, 4), 128, 0, stream>>>(in, Wsw, tb, N, Fin, Fout, mTiles);
    }
    int chunks = Fout >> 2;
    k_selfinit<<<cdiv((long long)N * chunks, 256), 256, 0, stream>>>(tb, dinv, ab, N, Fout);
    k_scatter<<<cdiv((long long)E * chunks, 256), 256, 0, stream>>>(src, dst, dinv, tb, ab, E, Fout);
    k_bias_relu<<<cdiv((long long)N * Fout, 256), 256, 0, stream>>>(ab, b, N, Fout);
  };

  layer(x,    W1, b1, t, bufA, F, F,  true);   // h1 in bufA
  layer(bufA, W2, b2, t, bufB, F, F,  true);   // h2 in bufB
  layer(bufB, W3, b3, t, bufA, F, F3, false);  // h3 in bufA (N x 8)

  k_final<<<cdiv((long long)N * C, 256), 256, 0, stream>>>(bufA, Wc, bc, (float*)d_out, N);
}
